// TanAILiteBlock_57775900066542
// MI455X (gfx1250) — matrix operand order repaired and checked by reading, not compiled, not run on hardware
//
#include <hip/hip_runtime.h>

// ---------------------------------------------------------------------------
// TanAI Lite transformer block for MI455X (gfx1250, wave32, WMMA)
// All GEMMs: bf16 inputs, fp32 accumulation via v_wmma_f32_16x16x32_bf16.
// Round 3: GEMM main loop unrolled 2x across the two LDS buffers so all
// ds_load offsets are static immediates; prefetch issued before WMMA chain.
// ---------------------------------------------------------------------------

#define D_MODEL   2048
#define MLP_DIM   8192
#define NHEADS    16
#define HEAD_DIM  128
#define SEQ       2048
#define BATCH     2
#define NTOK      (BATCH * SEQ)   // 4096

typedef __bf16 bf16;
typedef __attribute__((ext_vector_type(16))) __bf16 bf16x16;
typedef __attribute__((ext_vector_type(8)))  __bf16 bf16x8;
typedef __attribute__((ext_vector_type(8)))  float  f32x8;

union Frag { bf16x16 v; bf16x8 h[2]; };

// D = A(16x32) * B(32x16) + C, fp32 accum.
__device__ __forceinline__ f32x8 wmma_bf16(const Frag& a, const Frag& b, f32x8 c) {
  return __builtin_amdgcn_wmma_f32_16x16x32_bf16(false, a.v, false, b.v,
                                                 (short)0, c, false, false);
}

// ---------------------------------------------------------------------------
// Weight pre-pass: W[N,K] fp32 (row-major) -> WT[K,N] bf16 (row-major).
// WMMA's B operand wants row-major [K,N] (lane = N-col, K along elements, N contiguous).
// ---------------------------------------------------------------------------
__global__ __launch_bounds__(256)
void convert_transpose(const float* __restrict__ W, bf16* __restrict__ WT,
                       int Nr, int Kc) {
  __shared__ float tile[32][33];
  const int k0 = blockIdx.x * 32;
  const int n0 = blockIdx.y * 32;
  const int tx = threadIdx.x & 31;
  const int ty = threadIdx.x >> 5;           // 0..7
#pragma unroll
  for (int i = 0; i < 32; i += 8)
    tile[ty + i][tx] = W[(size_t)(n0 + ty + i) * Kc + k0 + tx];
  __syncthreads();
#pragma unroll
  for (int i = 0; i < 32; i += 8)
    WT[(size_t)(k0 + ty + i) * Nr + n0 + tx] = (bf16)tile[tx][ty + i];
}

// ---------------------------------------------------------------------------
// K-head transpose: kin[NTOK, D_MODEL] bf16 -> kT[B,H,Dh,S] bf16.
// Coalesced on both sides via 32x32 LDS tile. grid = (S/32, Dh/32, B*H).
// ---------------------------------------------------------------------------
__global__ __launch_bounds__(256)
void khead_transpose(const bf16* __restrict__ kin, bf16* __restrict__ kT) {
  __shared__ bf16 tile[32][34];
  const int st = blockIdx.x * 32;            // seq tile
  const int dt = blockIdx.y * 32;            // head-dim tile
  const int bh = blockIdx.z;                 // b*16 + h
  const int b  = bh >> 4, h = bh & 15;
  const int tx = threadIdx.x & 31, ty = threadIdx.x >> 5;
  const size_t ibase = (size_t)(b * SEQ + st) * D_MODEL + h * HEAD_DIM + dt;
#pragma unroll
  for (int i = 0; i < 32; i += 8)            // tile[s_local][d_local]
    tile[ty + i][tx] = kin[ibase + (size_t)(ty + i) * D_MODEL + tx];
  __syncthreads();
  const size_t obase = ((size_t)(bh * HEAD_DIM + dt)) * SEQ + st;
#pragma unroll
  for (int i = 0; i < 32; i += 8)            // kT[d][s], s contiguous
    kT[obase + (size_t)(ty + i) * SEQ + tx] = tile[tx][ty + i];
}

// ---------------------------------------------------------------------------
// RMSNorm: x[row,:D] fp32 -> out[row,:D] bf16. One block (8 waves) per row.
// ---------------------------------------------------------------------------
__global__ __launch_bounds__(256)
void rmsnorm_kernel(const float* __restrict__ x, const float* __restrict__ w,
                    bf16* __restrict__ out, int D) {
  const int row = blockIdx.x;
  const float* xr = x + (size_t)row * D;
  float ss = 0.f;
  for (int i = threadIdx.x; i < D; i += 256) { float v = xr[i]; ss += v * v; }
#pragma unroll
  for (int off = 16; off > 0; off >>= 1) ss += __shfl_xor(ss, off, 32);
  __shared__ float red[8];
  if ((threadIdx.x & 31) == 0) red[threadIdx.x >> 5] = ss;
  __syncthreads();
  float tot = 0.f;
#pragma unroll
  for (int i = 0; i < 8; ++i) tot += red[i];
  const float r = rsqrtf(tot / (float)D + 1e-6f);
  bf16* orow = out + (size_t)row * D;
  for (int i = threadIdx.x; i < D; i += 256)
    orow[i] = (bf16)(xr[i] * r * w[i]);
}

// ---------------------------------------------------------------------------
// GEMM: C[M,N] = A[M,K](bf16,rm) * B[K,N](bf16,rm), fp32 accum, WMMA.
// Block tile 128x128, BK=32, 256 threads = 8 waves (4x2), wave tile 32x64.
// Double-buffered LDS, loop unrolled 2x so buffer indices are constants.
// K must be a multiple of 64 (holds: 2048, 8192).
// Epilogue MODE: 0 = bf16 store
//               2 = fp32 store with residual add
//               3 = bf16 store with exact-erf GELU
// ---------------------------------------------------------------------------
template <int MODE>
__global__ __launch_bounds__(256)
void gemm128(const bf16* __restrict__ A, const bf16* __restrict__ B,
             int M, int N, int K,
             float* __restrict__ outF, bf16* __restrict__ outB,
             const float* __restrict__ res) {
  __shared__ bf16 As[2][128][40];    // 32 K-cols, pad->80B rows (16B multiple)
  __shared__ bf16 Bs[2][32][136];    // 128 N-cols, pad->272B rows

  const int t    = threadIdx.x;
  const int lane = t & 31;
  const int wave = t >> 5;
  const int wm   = wave >> 1;        // 0..3  -> 32 rows each
  const int wn   = wave & 1;         // 0..1  -> 64 cols each
  const int bm   = blockIdx.y * 128;
  const int bn   = blockIdx.x * 128;

  const int mq = lane & 15;
  const int kg = (lane >> 4) * 8;    // K sub-offset per A layout

  // per-thread staging slots (2 x 16B for A, 2 x 16B for B)
  const int idx0 = t, idx1 = t + 256;
  const int rA0 = idx0 >> 2, cA0 = (idx0 & 3) * 8;
  const int rA1 = idx1 >> 2, cA1 = (idx1 & 3) * 8;
  const int rB0 = idx0 >> 4, cB0 = (idx0 & 15) * 8;
  const int rB1 = idx1 >> 4, cB1 = (idx1 & 15) * 8;

  bf16x8 ga0, ga1, gb0, gb1;
  auto gload = [&](int k0) {
    ga0 = *(const bf16x8*)&A[(size_t)(bm + rA0) * K + k0 + cA0];
    ga1 = *(const bf16x8*)&A[(size_t)(bm + rA1) * K + k0 + cA1];
    gb0 = *(const bf16x8*)&B[(size_t)(k0 + rB0) * N + bn + cB0];
    gb1 = *(const bf16x8*)&B[(size_t)(k0 + rB1) * N + bn + cB1];
  };
  auto lstore = [&](int buf) {
    *(bf16x8*)&As[buf][rA0][cA0] = ga0;
    *(bf16x8*)&As[buf][rA1][cA1] = ga1;
    *(bf16x8*)&Bs[buf][rB0][cB0] = gb0;
    *(bf16x8*)&Bs[buf][rB1][cB1] = gb1;
  };

  f32x8 acc[2][4] = {};

  auto compute = [&](int buf) {      // called with literal buf -> static offsets
    Frag a[2], b[4];
#pragma unroll
    for (int i = 0; i < 2; ++i) {
      int r = wm * 32 + i * 16 + mq;
      a[i].h[0] = *(const bf16x8*)&As[buf][r][kg];
      a[i].h[1] = *(const bf16x8*)&As[buf][r][16 + kg];
    }
#pragma unroll
    for (int j = 0; j < 4; ++j) {
      int c = wn * 64 + j * 16;
#pragma unroll
      for (int s = 0; s < 8; ++s) {
        b[j].v[s]     = Bs[buf][kg + s][c + mq];
        b[j].v[8 + s] = Bs[buf][16 + kg + s][c + mq];
      }
    }
#pragma unroll
    for (int i = 0; i < 2; ++i)
#pragma unroll
      for (int j = 0; j < 4; ++j)
        acc[i][j] = wmma_bf16(a[i], b[j], acc[i][j]);
  };

  gload(0);
  lstore(0);
  __syncthreads();

  for (int k0 = 0; k0 < K; k0 += 64) {
    // ---- stage A: data for k0 is in buffer 0 ----
    gload(k0 + 32);                  // always in range (K % 64 == 0)
    compute(0);
    lstore(1);
    __syncthreads();

    // ---- stage B: data for k0+32 is in buffer 1 ----
    const bool more = (k0 + 64) < K;
    if (more) gload(k0 + 64);
    compute(1);
    if (more) lstore(0);
    __syncthreads();
  }

  // ---- epilogue (C/D layout: VGPR r -> row, lane&15 -> col) ----
#pragma unroll
  for (int i = 0; i < 2; ++i)
#pragma unroll
    for (int j = 0; j < 4; ++j) {
      const int n  = bn + wn * 64 + j * 16 + mq;
      const int mb = bm + wm * 32 + i * 16 + ((lane >> 4) * 8);
      f32x8 vv = acc[i][j];
#pragma unroll
      for (int r = 0; r < 8; ++r) {
        const int m = mb + r;
        const float val = vv[r];
        if constexpr (MODE == 0) {
          outB[(size_t)m * N + n] = (bf16)val;
        } else if constexpr (MODE == 2) {
          const size_t idx = (size_t)m * N + n;
          outF[idx] = res[idx] + val;
        } else {
          const float g = 0.5f * val * (1.0f + erff(val * 0.70710678118654752f));
          outB[(size_t)m * N + n] = (bf16)g;
        }
      }
    }
}

// ---------------------------------------------------------------------------
// Flash-style causal attention, one block per (b, h, 128-query tile).
// q,v: [NTOK, D_MODEL] bf16; kT: [B,H,Dh,S] bf16; att: [NTOK, D_MODEL] bf16.
// 8 waves; wave w owns 16 query rows. Online softmax over 64-key tiles.
// ---------------------------------------------------------------------------
__global__ __launch_bounds__(256)
void attention_kernel(const bf16* __restrict__ q, const bf16* __restrict__ kT,
                      const bf16* __restrict__ v, const int* __restrict__ amask,
                      bf16* __restrict__ att) {
  const int bid = blockIdx.x;
  const int qb = bid & 15;             // S/128 = 16 query tiles
  const int h  = (bid >> 4) & 15;
  const int b  = bid >> 8;

  __shared__ bf16  Qs[128][HEAD_DIM + 8];   // 34.8 KB
  __shared__ bf16  Ks[HEAD_DIM][64 + 8];    // 18.4 KB  [d][key]
  __shared__ bf16  Vs[64][HEAD_DIM + 8];    // 17.4 KB  [key][d]
  __shared__ float Sc[128][64 + 1];         // 33.3 KB  scores fp32
  __shared__ bf16  Ps[128][64 + 8];         // 18.4 KB  probs (A operand)
  __shared__ float row_m[128], row_l[128], row_sc[128];
  __shared__ int   km[64];

  const int t    = threadIdx.x;
  const int lane = t & 31;
  const int wave = t >> 5;
  const int mq   = lane & 15;
  const int kg   = (lane >> 4) * 8;
  const int qrow0 = wave * 16;

  // load Q tile (rows = tokens, cols = this head's 128 dims)
  const size_t qbase = (size_t)(b * SEQ + qb * 128) * D_MODEL + h * HEAD_DIM;
  for (int c = t; c < 128 * 16; c += 256) {
    int r = c >> 4, col = (c & 15) * 8;
    *(bf16x8*)&Qs[r][col] = *(const bf16x8*)&q[qbase + (size_t)r * D_MODEL + col];
  }
  if (t < 128) { row_m[t] = -3.0e38f; row_l[t] = 0.f; }

  f32x8 accO[8] = {};
  const int nj = (qb + 1) * 2;         // 64-key blocks up to & incl. diagonal

  for (int j = 0; j < nj; ++j) {
    const int j0 = j * 64;
    __syncthreads();   // prior consumers of Ks/Vs/Ps done (covers Q load too)

    const size_t kbase = ((size_t)(b * 16 + h) * HEAD_DIM) * SEQ + j0;
    for (int c = t; c < 128 * 8; c += 256) {           // Ks: [d][key]
      int d = c >> 3, col = (c & 7) * 8;
      *(bf16x8*)&Ks[d][col] = *(const bf16x8*)&kT[kbase + (size_t)d * SEQ + col];
    }
    const size_t vbase = (size_t)(b * SEQ + j0) * D_MODEL + h * HEAD_DIM;
    for (int c = t; c < 64 * 16; c += 256) {           // Vs: [key][d]
      int r = c >> 4, col = (c & 15) * 8;
      *(bf16x8*)&Vs[r][col] = *(const bf16x8*)&v[vbase + (size_t)r * D_MODEL + col];
    }
    if (t < 64) km[t] = amask[b * SEQ + j0 + t];
    __syncthreads();

    // ---- scores = Q * K^T (wave: 16 q-rows x 64 keys) ----
    f32x8 accS[4] = {};
#pragma unroll
    for (int kk = 0; kk < 4; ++kk) {
      Frag a;
      const int r = qrow0 + mq;
      a.h[0] = *(const bf16x8*)&Qs[r][kk * 32 + kg];
      a.h[1] = *(const bf16x8*)&Qs[r][kk * 32 + 16 + kg];
#pragma unroll
      for (int nt = 0; nt < 4; ++nt) {
        Frag bb;
#pragma unroll
        for (int s = 0; s < 8; ++s) {
          bb.v[s]     = Ks[kk * 32 + kg + s][nt * 16 + mq];
          bb.v[8 + s] = Ks[kk * 32 + 16 + kg + s][nt * 16 + mq];
        }
        accS[nt] = wmma_bf16(a, bb, accS[nt]);
      }
    }
    // scale + causal/key mask, write fp32 scores to LDS
    const float sscale = 0.088388347648318447f;   // 1/sqrt(128)
    const int qg0   = qb * 128 + qrow0 + (lane >> 4) * 8;
    const int mrow0 = qrow0 + (lane >> 4) * 8;
#pragma unroll
    for (int nt = 0; nt < 4; ++nt) {
      const int key_l = nt * 16 + mq;
      const int keyg  = j0 + key_l;
      const bool kok  = (km[key_l] != 0);
#pragma unroll
      for (int r = 0; r < 8; ++r) {
        float s = accS[nt][r] * sscale;
        if ((keyg > qg0 + r) || !kok) s = -1e9f;
        Sc[mrow0 + r][key_l] = s;
      }
    }
    __syncthreads();

    // ---- online softmax, one thread per query row ----
    if (t < 128) {
      const float mo = row_m[t];
      float mn = mo;
      for (int ki = 0; ki < 64; ++ki) mn = fmaxf(mn, Sc[t][ki]);
      const float scl = __expf(mo - mn);
      float sum = 0.f;
      for (int ki = 0; ki < 64; ++ki) {
        const float sv = Sc[t][ki];
        const float p  = (sv <= -1e8f) ? 0.f : __expf(sv - mn);
        Ps[t][ki] = (bf16)p;
        sum += p;
      }
      row_m[t]  = mn;
      row_l[t]  = row_l[t] * scl + sum;
      row_sc[t] = scl;
    }
    __syncthreads();

    // ---- rescale O, then O += P * V ----
    {
      float rs[8];
      const int base = qrow0 + (lane >> 4) * 8;
#pragma unroll
      for (int r = 0; r < 8; ++r) rs[r] = row_sc[base + r];
#pragma unroll
      for (int d8 = 0; d8 < 8; ++d8)
#pragma unroll
        for (int r = 0; r < 8; ++r) accO[d8][r] *= rs[r];
#pragma unroll
      for (int kk = 0; kk < 2; ++kk) {
        Frag a;
        const int r = qrow0 + mq;
        a.h[0] = *(const bf16x8*)&Ps[r][kk * 32 + kg];
        a.h[1] = *(const bf16x8*)&Ps[r][kk * 32 + 16 + kg];
#pragma unroll
        for (int nt = 0; nt < 8; ++nt) {
          Frag bb;
#pragma unroll
          for (int s = 0; s < 8; ++s) {
            bb.v[s]     = Vs[kk * 32 + kg + s][nt * 16 + mq];
            bb.v[8 + s] = Vs[kk * 32 + 16 + kg + s][nt * 16 + mq];
          }
          accO[nt] = wmma_bf16(a, bb, accO[nt]);
        }
      }
    }
  }

  // ---- normalize by row sums and store att[token, h*128+d] bf16 ----
  {
    const int base = qrow0 + (lane >> 4) * 8;
    float inv[8];
#pragma unroll
    for (int r = 0; r < 8; ++r) {
      const float l = row_l[base + r];
      inv[r] = (l > 0.f) ? 1.f / l : 0.f;
    }
    const size_t obase = (size_t)(b * SEQ + qb * 128) * D_MODEL + h * HEAD_DIM;
#pragma unroll
    for (int nt = 0; nt < 8; ++nt) {
      const int d = nt * 16 + mq;
#pragma unroll
      for (int r = 0; r < 8; ++r)
        att[obase + (size_t)(base + r) * D_MODEL + d] = (bf16)(accO[nt][r] * inv[r]);
    }
  }
}

// ---------------------------------------------------------------------------
// Launch
// ---------------------------------------------------------------------------
extern "C" void kernel_launch(void* const* d_in, const int* in_sizes, int n_in,
                              void* d_out, int out_size, void* d_ws, size_t ws_size,
                              hipStream_t stream) {
  (void)in_sizes; (void)n_in; (void)out_size; (void)ws_size;
  const float* x      = (const float*)d_in[0];
  const int*   amask  = (const int*)d_in[1];
  const float* wna    = (const float*)d_in[2];
  const float* wq     = (const float*)d_in[3];
  const float* wk     = (const float*)d_in[4];
  const float* wv     = (const float*)d_in[5];
  const float* wo     = (const float*)d_in[6];
  const float* wnm    = (const float*)d_in[7];
  const float* w_up   = (const float*)d_in[8];
  const float* w_down = (const float*)d_in[9];
  float* out = (float*)d_out;

  size_t off = 0;
  auto walloc = [&](size_t bytes) {
    void* p = (char*)d_ws + off;
    off += (bytes + 255) & ~(size_t)255;
    return p;
  };
  const size_t DD  = (size_t)D_MODEL * D_MODEL;
  const size_t DM  = (size_t)D_MODEL * MLP_DIM;
  const size_t TD  = (size_t)NTOK * D_MODEL;
  const size_t TM  = (size_t)NTOK * MLP_DIM;

  bf16* wqT   = (bf16*)walloc(DD * 2);
  bf16* wkT   = (bf16*)walloc(DD * 2);
  bf16* wvT   = (bf16*)walloc(DD * 2);
  bf16* woT   = (bf16*)walloc(DD * 2);
  bf16* wupT  = (bf16*)walloc(DM * 2);   // [D, MLP]
  bf16* wdnT  = (bf16*)walloc(DM * 2);   // [MLP, D]
  bf16* h1    = (bf16*)walloc(TD * 2);
  bf16* qbuf  = (bf16*)walloc(TD * 2);
  bf16* kbuf  = (bf16*)walloc(TD * 2);   // [token, D]
  bf16* ktbuf = (bf16*)walloc(TD * 2);   // [B,H,Dh,S]
  bf16* vbuf  = (bf16*)walloc(TD * 2);
  bf16* attb  = (bf16*)walloc(TD * 2);
  float* x1   = (float*)walloc(TD * 4);
  bf16* h2    = (bf16*)walloc(TD * 2);
  bf16* up    = (bf16*)walloc(TM * 2);

  const dim3 t256(256);

  // weights -> bf16, transposed to [K,N]
  convert_transpose<<<dim3(64, 64),  t256, 0, stream>>>(wq,     wqT,  D_MODEL, D_MODEL);
  convert_transpose<<<dim3(64, 64),  t256, 0, stream>>>(wk,     wkT,  D_MODEL, D_MODEL);
  convert_transpose<<<dim3(64, 64),  t256, 0, stream>>>(wv,     wvT,  D_MODEL, D_MODEL);
  convert_transpose<<<dim3(64, 64),  t256, 0, stream>>>(wo,     woT,  D_MODEL, D_MODEL);
  convert_transpose<<<dim3(64, 256), t256, 0, stream>>>(w_up,   wupT, MLP_DIM, D_MODEL);
  convert_transpose<<<dim3(256, 64), t256, 0, stream>>>(w_down, wdnT, D_MODEL, MLP_DIM);

  // attention half
  rmsnorm_kernel<<<NTOK, t256, 0, stream>>>(x, wna, h1, D_MODEL);
  gemm128<0><<<dim3(16, 32), t256, 0, stream>>>(h1, wqT, NTOK, D_MODEL, D_MODEL, nullptr, qbuf, nullptr);
  gemm128<0><<<dim3(16, 32), t256, 0, stream>>>(h1, wkT, NTOK, D_MODEL, D_MODEL, nullptr, kbuf, nullptr);
  gemm128<0><<<dim3(16, 32), t256, 0, stream>>>(h1, wvT, NTOK, D_MODEL, D_MODEL, nullptr, vbuf, nullptr);
  khead_transpose<<<dim3(SEQ / 32, HEAD_DIM / 32, BATCH * NHEADS), t256, 0, stream>>>(kbuf, ktbuf);
  attention_kernel<<<BATCH * NHEADS * (SEQ / 128), t256, 0, stream>>>(qbuf, ktbuf, vbuf, amask, attb);
  gemm128<2><<<dim3(16, 32), t256, 0, stream>>>(attb, woT, NTOK, D_MODEL, D_MODEL, x1, nullptr, x);

  // MLP half
  rmsnorm_kernel<<<NTOK, t256, 0, stream>>>(x1, wnm, h2, D_MODEL);
  gemm128<3><<<dim3(64, 32), t256, 0, stream>>>(h2, wupT, NTOK, MLP_DIM, D_MODEL, nullptr, up, nullptr);
  gemm128<2><<<dim3(16, 32), t256, 0, stream>>>(up, wdnT, NTOK, D_MODEL, MLP_DIM, out, nullptr, x1);
}
